// EFQ_62070867362445
// MI455X (gfx1250) — matrix operand order repaired and checked by reading, not compiled, not run on hardware
//
#include <hip/hip_runtime.h>

// ---------------------------------------------------------------------------
// 8-qubit VQC, batch 16384.  Final = S @ U^T  (complex), done as 4 real f16
// WMMA GEMMs with f32 accumulation; measurement fused into the epilogue.
// Bit convention (matches reference): wire i  <->  bit (7-i) of flat index.
//   Kernel 1: build the 256x256 variational unitary (one wave per column).
//   Kernel 2: wave-pairs own a 16-row batch slab; each wave does 8 of the
//             16 N-chunks; signed |.|^2 reduction merges via LDS f32 atomics.
// ---------------------------------------------------------------------------

typedef __attribute__((ext_vector_type(16))) _Float16     v16h;
typedef __attribute__((ext_vector_type(8)))  float        v8f;
typedef __attribute__((ext_vector_type(8)))  unsigned int v8u;

struct c32 { float x, y; };
__device__ __forceinline__ c32 cmul(c32 a, c32 b) {
    return { a.x*b.x - a.y*b.y, a.x*b.y + a.y*b.x };
}
__device__ __forceinline__ c32 cadd(c32 a, c32 b) { return { a.x+b.x, a.y+b.y }; }

__device__ __forceinline__ float shflx(float v, int mask) {
    return __shfl_xor(v, mask, 32);
}

// Apply a 2x2 complex gate on bit position p of the 256-amplitude state held
// as 8 complex values per lane (idx = lane*8 + j; bits[2:0]=j, bits[7:3]=lane).
__device__ __forceinline__ void apply2(c32 (&amp)[8], int lane, int p,
                                       c32 u00, c32 u01, c32 u10, c32 u11) {
    if (p >= 3) {                       // paired across lanes
        const int mask = 1 << (p - 3);
        const bool hi = (lane & mask) != 0;
        #pragma unroll
        for (int j = 0; j < 8; ++j) {
            c32 part = { shflx(amp[j].x, mask), shflx(amp[j].y, mask) };
            c32 a0 = hi ? part : amp[j];
            c32 a1 = hi ? amp[j] : part;
            c32 r0 = cadd(cmul(u00, a0), cmul(u01, a1));
            c32 r1 = cadd(cmul(u10, a0), cmul(u11, a1));
            amp[j] = hi ? r1 : r0;
        }
    } else {                            // paired inside the lane
        const int mb = 1 << p;
        #pragma unroll
        for (int j = 0; j < 8; ++j) {
            if (j & mb) continue;
            const int j1 = j | mb;
            c32 a0 = amp[j], a1 = amp[j1];
            amp[j]  = cadd(cmul(u00, a0), cmul(u01, a1));
            amp[j1] = cadd(cmul(u10, a0), cmul(u11, a1));
        }
    }
}

// CNOT: control wire q (bit 7-q), target wire q+1 (bit 6-q).
__device__ __forceinline__ void cnot_gate(c32 (&amp)[8], int lane, int q) {
    const int pc = 7 - q, pt = 6 - q;
    if (pt >= 3) {                      // ctrl & tgt both lane bits (q<=3)
        const int cm = 1 << (pc - 3), tm = 1 << (pt - 3);
        const bool ctl = (lane & cm) != 0;
        #pragma unroll
        for (int j = 0; j < 8; ++j) {
            c32 part = { shflx(amp[j].x, tm), shflx(amp[j].y, tm) };
            amp[j] = ctl ? part : amp[j];
        }
    } else if (pc >= 3) {               // ctrl lane bit, tgt local (q==4)
        const int cm = 1 << (pc - 3), tm = 1 << pt;
        const bool ctl = (lane & cm) != 0;
        #pragma unroll
        for (int j = 0; j < 8; ++j) {
            if (j & tm) continue;
            const int j1 = j | tm;
            c32 a = amp[j], b = amp[j1];
            amp[j]  = ctl ? b : a;
            amp[j1] = ctl ? a : b;
        }
    } else {                            // both local (q>=5)
        const int cmb = 1 << pc, tmb = 1 << pt;
        #pragma unroll
        for (int j = 0; j < 8; ++j) {
            if (!(j & cmb) || (j & tmb)) continue;
            const int j1 = j | tmb;
            c32 t = amp[j]; amp[j] = amp[j1]; amp[j1] = t;
        }
    }
}

// ---------------------------------------------------------------------------
// Kernel 1: build variational unitary U (256x256 complex) as f16 planes
// U[j][k] row-major: Ure, Uim.  One wave per basis column k.
// ---------------------------------------------------------------------------
__global__ __launch_bounds__(32)
void build_unitary(const float* __restrict__ w,
                   _Float16* __restrict__ Ure,
                   _Float16* __restrict__ Uim) {
    const int k    = blockIdx.x;
    const int lane = threadIdx.x;

    c32 amp[8];
    #pragma unroll
    for (int j = 0; j < 8; ++j) amp[j] = { 0.f, 0.f };
    if ((k >> 3) == lane) amp[k & 7] = { 1.f, 0.f };

    int idx = 0;
    #pragma unroll
    for (int layer = 0; layer < 3; ++layer) {
        #pragma unroll
        for (int q = 0; q < 8; ++q) {
            const int p  = 7 - q;
            const float hx = w[idx] * 0.5f, hy = w[idx + 1] * 0.5f; idx += 2;
            const float cx = __cosf(hx), sx = __sinf(hx);
            const float cy = __cosf(hy), sy = __sinf(hy);
            apply2(amp, lane, p, {cx,0.f}, {0.f,-sx}, {0.f,-sx}, {cx,0.f});  // RX
            apply2(amp, lane, p, {cy,0.f}, {-sy,0.f}, {sy,0.f},  {cy,0.f});  // RY
        }
        #pragma unroll
        for (int q = 0; q < 7; ++q) cnot_gate(amp, lane, q);
    }
    #pragma unroll
    for (int j = 0; j < 8; ++j) {
        const int row = (lane << 3) | j;
        const size_t o = (size_t)row * 256 + k;
        Ure[o] = (_Float16)amp[j].x;
        Uim[o] = (_Float16)amp[j].y;
    }
}

// ---------------------------------------------------------------------------
// Kernel 2: fused encode -> complex WMMA GEMM -> |.|^2 -> signed reduction.
// 8 waves / block; wave-pair (wave>>1) owns a 16-row slab; (wave&1) selects
// which 8 of the 16 column chunks this wave computes.
// ---------------------------------------------------------------------------
__global__ __launch_bounds__(256)
void qgemm(const float* __restrict__ x,
           const _Float16* __restrict__ Ure,
           const _Float16* __restrict__ Uim,
           float* __restrict__ out) {
    __shared__ float lds_out[4][16][8];          // [row-slab][m][qubit]
    const int lane = threadIdx.x & 31;
    const int wave = threadIdx.x >> 5;
    const int slab = wave >> 1;
    const int row0 = (blockIdx.x * 4 + slab) * 16;
    const int c0   = (wave & 1) * 8;             // this wave's chunk range

    // zero the block's 512 accumulation floats, then fence (pairs share slabs)
    ((float*)lds_out)[threadIdx.x]       = 0.f;
    ((float*)lds_out)[threadIdx.x + 256] = 0.f;
    __syncthreads();

    // ---- per-lane rank-1 encoding (A row: M = lane&15, this lane's K half)
    const int b = row0 + (lane & 15);
    const float4 x0 = ((const float4*)(x + (size_t)b * 8))[0];
    const float4 x1 = ((const float4*)(x + (size_t)b * 8))[1];
    const float xs[8] = { x0.x, x0.y, x0.z, x0.w, x1.x, x1.y, x1.z, x1.w };
    c32 e0[8], e1[8];                       // v_i = RZ(a)RY(a)|0>, a = x*pi
    #pragma unroll
    for (int i = 0; i < 8; ++i) {
        const float h = xs[i] * 1.57079632679489662f;   // a/2
        const float c = __cosf(h), s = __sinf(h);
        e0[i] = { c * c, -s * c };
        e1[i] = { s * c,  s * s };
    }
    c32 low8[8];                            // wires 5,6,7 (bits 2..0)
    #pragma unroll
    for (int t = 0; t < 8; ++t)
        low8[t] = cmul(cmul((t & 4) ? e1[5] : e0[5],
                            (t & 2) ? e1[6] : e0[6]),
                            (t & 1) ? e1[7] : e0[7]);
    const c32 F = (lane >= 16) ? e1[3] : e0[3];   // bit 4 fixed per lane half

    // A fragments in contiguous K order (same as B): elem e -> K=f*32+off+e,
    // off = 16*(lane>=16).
    v16h Ar[8], Ai[8];
    #pragma unroll
    for (int f = 0; f < 8; ++f) {
        c32 g = (f & 4) ? e1[0] : e0[0];
        g = cmul(g, (f & 2) ? e1[1] : e0[1]);
        g = cmul(g, (f & 1) ? e1[2] : e0[2]);
        g = cmul(g, F);
        #pragma unroll
        for (int gg = 0; gg < 2; ++gg) {
            const c32 H = cmul(g, gg ? e1[4] : e0[4]);
            #pragma unroll
            for (int t = 0; t < 8; ++t) {
                const c32 a = cmul(H, low8[t]);
                Ar[f][gg * 8 + t] = (_Float16)a.x;
                Ai[f][gg * 8 + t] = (_Float16)a.y;
            }
        }
    }

    float P[8];                 // plain prob sum  (signs from n-bits, qubits 4..7)
    float Q[4][8];              // chunk-signed sums        (qubits 0..3)
    #pragma unroll
    for (int r = 0; r < 8; ++r) { P[r] = 0.f; Q[0][r] = Q[1][r] = Q[2][r] = Q[3][r] = 0.f; }

    const int n    = lane & 15;
    const int koff = (lane >> 4) << 4;      // B lanes 16..31 hold K+16..K+31
    const v8u sgn  = (v8u)(0x80008000u);    // f16-pair sign-flip mask

    for (int cc = 0; cc < 8; ++cc) {        // this wave's 8 column chunks
        const int c = c0 + cc;
        v8f acc_re = {}, acc_im = {};
        const int j = c * 16 + n;
        const _Float16* pre = Ure + (size_t)j * 256 + koff;
        const _Float16* pim = Uim + (size_t)j * 256 + koff;
        #pragma unroll
        for (int s = 0; s < 8; ++s) {       // K = 256 in steps of 32
            const v16h Br  = *(const v16h*)(pre + s * 32);
            const v16h Bi  = *(const v16h*)(pim + s * 32);
            const v16h Bin = __builtin_bit_cast(v16h, __builtin_bit_cast(v8u, Bi) ^ sgn);
            // Re += Ar*Br - Ai*Bi ;  Im += Ar*Bi + Ai*Br
            acc_re = __builtin_amdgcn_wmma_f32_16x16x32_f16(false, Ar[s], false, Br,  (short)0, acc_re, false, false);
            acc_re = __builtin_amdgcn_wmma_f32_16x16x32_f16(false, Ai[s], false, Bin, (short)0, acc_re, false, false);
            acc_im = __builtin_amdgcn_wmma_f32_16x16x32_f16(false, Ar[s], false, Bi,  (short)0, acc_im, false, false);
            acc_im = __builtin_amdgcn_wmma_f32_16x16x32_f16(false, Ai[s], false, Br,  (short)0, acc_im, false, false);
        }
        #pragma unroll
        for (int r = 0; r < 8; ++r) {       // C layout: this lane = col n, rows r(+8)
            const float re = acc_re[r], im = acc_im[r];
            const float p = re * re + im * im;
            P[r] += p;
            #pragma unroll
            for (int i = 0; i < 4; ++i)     // qubit i <-> bit (3-i) of chunk c
                Q[i][r] += ((c >> (3 - i)) & 1) ? -p : p;
        }
    }

    // signed reduction over columns: merge both waves of the pair via LDS
    const int mbase = (lane >> 4) * 8;
    #pragma unroll
    for (int r = 0; r < 8; ++r) {
        const int m = mbase + r;
        #pragma unroll
        for (int i = 0; i < 4; ++i)
            atomicAdd(&lds_out[slab][m][i], Q[i][r]);
        #pragma unroll
        for (int i = 4; i < 8; ++i)         // qubit i <-> bit (7-i) of n
            atomicAdd(&lds_out[slab][m][i], ((n >> (7 - i)) & 1) ? -P[r] : P[r]);
    }
    __syncthreads();
    #pragma unroll
    for (int t2 = 0; t2 < 2; ++t2) {        // 512 outputs per block
        const int u  = threadIdx.x + 256 * t2;
        const int sl = u >> 7;
        const int m  = (u >> 3) & 15;
        const int i  = u & 7;
        out[(size_t)(blockIdx.x * 64 + sl * 16 + m) * 8 + i] = lds_out[sl][m][i];
    }
}

// ---------------------------------------------------------------------------
extern "C" void kernel_launch(void* const* d_in, const int* in_sizes, int n_in,
                              void* d_out, int out_size, void* d_ws, size_t ws_size,
                              hipStream_t stream) {
    const float* x = (const float*)d_in[0];     // [B,8] f32
    const float* w = (const float*)d_in[1];     // [48]  f32
    float* out = (float*)d_out;                 // [B,8] f32

    char* ws = (char*)d_ws;                     // 2 x 128KB f16 planes
    _Float16* Ure = (_Float16*)(ws);
    _Float16* Uim = (_Float16*)(ws + 128 * 1024);

    const int B = in_sizes[0] / 8;              // 16384

    build_unitary<<<256, 32, 0, stream>>>(w, Ure, Uim);
    qgemm<<<B / 64, 256, 0, stream>>>(x, Ure, Uim, out);
}
